// PairwiseAttentionLayer_4209067950293
// MI455X (gfx1250) — matrix operand order repaired and checked by reading, not compiled, not run on hardware
//
#include <hip/hip_runtime.h>

#define B_DIM 4
#define T_DIM 512
#define H_DIM 512
#define U_DIM 256

typedef __attribute__((ext_vector_type(16))) __bf16 v16bf;
typedef __attribute__((ext_vector_type(8)))  __bf16 v8bf;
typedef __attribute__((ext_vector_type(8)))  float  v8f;

// Hardware tanh (CDNA5 TRANS op v_tanh_f32) if the toolchain exposes it.
#if __has_builtin(__builtin_amdgcn_tanhf)
__device__ __forceinline__ float fast_tanh(float x) { return __builtin_amdgcn_tanhf(x); }
#elif __has_builtin(__builtin_amdgcn_tanh_f32)
__device__ __forceinline__ float fast_tanh(float x) { return __builtin_amdgcn_tanh_f32(x); }
#else
__device__ __forceinline__ float fast_tanh(float x) { return tanhf(x); }
#endif

#if __has_builtin(__builtin_amdgcn_rcpf)
__device__ __forceinline__ float fast_rcp(float x) { return __builtin_amdgcn_rcpf(x); }
#else
__device__ __forceinline__ float fast_rcp(float x) { return 1.f / x; }
#endif

// ---------------- fp32 -> bf16 convert ----------------
__global__ void cvt_f32_bf16(const float* __restrict__ src, __bf16* __restrict__ dst, int n) {
    int i = blockIdx.x * blockDim.x + threadIdx.x;
    if (i < n) dst[i] = (__bf16)src[i];
}

__global__ void zero_f32(float* __restrict__ p, int n) {
    int i = blockIdx.x * blockDim.x + threadIdx.x;
    if (i < n) p[i] = 0.f;
}

// ---------------- WMMA projection GEMM ----------------
// C[M x N] = A[M x K] * W[K x N], row-major, bf16 in / f32 out.
// One wave (32 threads) per 16x16 C tile; K-loop in steps of 32.
// grid = (N/16, M/16, 2); z selects (Wq -> Hq) or (Wk -> Hk).
__global__ void proj_gemm_wmma(const __bf16* __restrict__ A,
                               const __bf16* __restrict__ Wq,
                               const __bf16* __restrict__ Wk,
                               float* __restrict__ Hq, float* __restrict__ Hk,
                               int N, int K) {
    const __bf16* Bm = (blockIdx.z == 0) ? Wq : Wk;
    float*        Cm = (blockIdx.z == 0) ? Hq : Hk;

    const int lane = threadIdx.x;          // wave32
    const int m0 = blockIdx.y * 16;
    const int n0 = blockIdx.x * 16;
    const int half = lane >> 4;            // 0..1
    const int mrow = lane & 15;

    union { v16bf v; v8bf h[2]; } a, b;
    v8f c = {};

    for (int k0 = 0; k0 < K; k0 += 32) {
        // A 16x32 bf16 fragment (ISA 7.12.2): lane holds row m=lane%16;
        // dwords 0..3 = K {half*8 .. half*8+7}, dwords 4..7 = K {16+half*8 ..}.
        const __bf16* ap = A + (size_t)(m0 + mrow) * K + k0 + half * 8;
        a.h[0] = *(const v8bf*)(ap);
        a.h[1] = *(const v8bf*)(ap + 16);
        // B 32x16 bf16 fragment: lane holds col N = lane%16; K same order as A.
        const __bf16* bp = Bm + (size_t)(k0 + half * 8) * N + n0 + mrow;
#pragma unroll
        for (int s = 0; s < 8; ++s) {
            b.v[s]     = bp[(size_t)s * N];
            b.v[8 + s] = bp[(size_t)(16 + s) * N];
        }
        // D = A*B + C  ->  v_wmma_f32_16x16x32_bf16
        c = __builtin_amdgcn_wmma_f32_16x16x32_bf16(false, a.v, false, b.v,
                                                    (short)0, c, false, false);
    }
    // C/D layout: VGPR r -> row r + 8*(lane/16), col lane%16.
#pragma unroll
    for (int r = 0; r < 8; ++r) {
        Cm[(size_t)(m0 + r + half * 8) * N + n0 + mrow] = c[r];
    }
}

// ---------------- pairwise score + softmax + beta ----------------
// One 256-thread block per (b, t). Wave-per-s, lanes split u (float4).
__global__ void score_softmax_beta(const float* __restrict__ Hq,
                                   const float* __restrict__ Hk,
                                   const float* __restrict__ bq,
                                   const float* __restrict__ Wv,
                                   const float* __restrict__ balpha,
                                   float* __restrict__ beta) {
    __shared__ __align__(16) float s_hq[U_DIM];
    __shared__ __align__(16) float s_w[U_DIM];
    __shared__ float s_score[T_DIM];
    __shared__ float s_red[8];

    const int tid = threadIdx.x;
    const int b = blockIdx.x / T_DIM;
    const int t = blockIdx.x % T_DIM;
    const int lane = tid & 31, wave = tid >> 5;   // 8 waves

    s_hq[tid] = Hq[((size_t)b * T_DIM + t) * U_DIM + tid] + bq[tid];
    s_w[tid]  = Wv[tid];
    __syncthreads();

    const float ba = balpha[0];
    for (int s = wave; s < T_DIM; s += 8) {
        const float* hkrow = Hk + ((size_t)b * T_DIM + s) * U_DIM;
        if (s + 8 < T_DIM)  // speculative prefetch of next row (global_prefetch_b8)
            __builtin_prefetch(hkrow + 8 * U_DIM + lane * 8, 0, 0);
        float acc = 0.f;
#pragma unroll
        for (int ch = 0; ch < 2; ++ch) {
            const int u0 = ch * 128 + lane * 4;
            float4 k4 = *(const float4*)(hkrow + u0);
            float4 q4 = *(const float4*)(s_hq + u0);
            float4 w4 = *(const float4*)(s_w + u0);
            acc += fast_tanh(q4.x + k4.x) * w4.x;
            acc += fast_tanh(q4.y + k4.y) * w4.y;
            acc += fast_tanh(q4.z + k4.z) * w4.z;
            acc += fast_tanh(q4.w + k4.w) * w4.w;
        }
#pragma unroll
        for (int off = 16; off > 0; off >>= 1) acc += __shfl_xor(acc, off, 32);
        if (lane == 0) {
            // sigmoid via hardware exp + rcp
            s_score[s] = fast_rcp(1.f + __expf(-(acc + ba)));
        }
    }
    __syncthreads();

    // softmax over s (512 values, 2 per thread)
    float v0 = s_score[tid], v1 = s_score[tid + 256];
    float m = fmaxf(v0, v1);
#pragma unroll
    for (int off = 16; off > 0; off >>= 1) m = fmaxf(m, __shfl_xor(m, off, 32));
    if (lane == 0) s_red[wave] = m;
    __syncthreads();
    if (tid == 0) {
        float mm = s_red[0];
#pragma unroll
        for (int i = 1; i < 8; ++i) mm = fmaxf(mm, s_red[i]);
        s_red[0] = mm;
    }
    __syncthreads();
    m = s_red[0];
    __syncthreads();

    float e0 = __expf(v0 - m), e1 = __expf(v1 - m);
    float sm = e0 + e1;
#pragma unroll
    for (int off = 16; off > 0; off >>= 1) sm += __shfl_xor(sm, off, 32);
    if (lane == 0) s_red[wave] = sm;
    __syncthreads();
    if (tid == 0) {
        float ss = 0.f;
#pragma unroll
        for (int i = 0; i < 8; ++i) ss += s_red[i];
        s_red[0] = ss;
    }
    __syncthreads();
    const float scale = fast_rcp(s_red[0]) * (1.0f / (float)T_DIM);

    // beta[b,s] = (1/T) * sum_t alpha[b,t,s]  (mean over t folded in)
    atomicAdd(&beta[(size_t)b * T_DIM + tid],       e0 * scale);
    atomicAdd(&beta[(size_t)b * T_DIM + tid + 256], e1 * scale);
}

// ---------------- context[b,:] = sum_s beta[b,s] * h[b,s,:] ----------------
__global__ void context_kernel(const float* __restrict__ h,
                               const float* __restrict__ beta,
                               float* __restrict__ out) {
    __shared__ float s_beta[T_DIM];
    const int b = blockIdx.x;
    const int tid = threadIdx.x;  // 512 = H_DIM
    s_beta[tid] = beta[(size_t)b * T_DIM + tid];
    __syncthreads();
    const float* hb = h + (size_t)b * T_DIM * H_DIM;
    float acc = 0.f;
    for (int s = 0; s < T_DIM; ++s)
        acc = fmaf(s_beta[s], hb[(size_t)s * H_DIM + tid], acc);
    out[(size_t)b * H_DIM + tid] = acc;
}

extern "C" void kernel_launch(void* const* d_in, const int* in_sizes, int n_in,
                              void* d_out, int out_size, void* d_ws, size_t ws_size,
                              hipStream_t stream) {
    const float* h  = (const float*)d_in[0];   // (B,T,H)
    const float* Wq = (const float*)d_in[1];   // (H,U)
    const float* Wk = (const float*)d_in[2];   // (H,U)
    const float* bq = (const float*)d_in[3];   // (U,)
    const float* Wv = (const float*)d_in[4];   // (U,1)
    const float* ba = (const float*)d_in[5];   // (1,)
    float* out = (float*)d_out;                // (B,H)

    // workspace layout (bytes), ~6.6 MB total
    char* ws = (char*)d_ws;
    __bf16* h_bf  = (__bf16*)(ws + 0);          // 2 MB
    __bf16* wq_bf = (__bf16*)(ws + 2097152);    // 256 KB
    __bf16* wk_bf = (__bf16*)(ws + 2359296);    // 256 KB
    float*  hq    = (float*)(ws + 2621440);     // 2 MB
    float*  hk    = (float*)(ws + 4718592);     // 2 MB
    float*  beta  = (float*)(ws + 6815744);     // 8 KB

    const int nh = B_DIM * T_DIM * H_DIM;   // 1048576
    const int nw = H_DIM * U_DIM;           // 131072
    cvt_f32_bf16<<<(nh + 255) / 256, 256, 0, stream>>>(h,  h_bf,  nh);
    cvt_f32_bf16<<<(nw + 255) / 256, 256, 0, stream>>>(Wq, wq_bf, nw);
    cvt_f32_bf16<<<(nw + 255) / 256, 256, 0, stream>>>(Wk, wk_bf, nw);
    zero_f32<<<(B_DIM * T_DIM + 255) / 256, 256, 0, stream>>>(beta, B_DIM * T_DIM);

    dim3 ggrid(U_DIM / 16, (B_DIM * T_DIM) / 16, 2);  // (16,128,2) tiles
    proj_gemm_wmma<<<ggrid, 32, 0, stream>>>(h_bf, wq_bf, wk_bf, hq, hk, U_DIM, H_DIM);

    score_softmax_beta<<<B_DIM * T_DIM, 256, 0, stream>>>(hq, hk, bq, Wv, ba, beta);

    context_kernel<<<B_DIM, H_DIM, 0, stream>>>(h, beta, out);
}
